// MultiScaleMemoryLayerParallel_57844619542835
// MI455X (gfx1250) — matrix operand order repaired and checked by reading, not compiled, not run on hardware
//
#include <hip/hip_runtime.h>
#include <hip/hip_bf16.h>

typedef __attribute__((ext_vector_type(16))) __bf16 v16bf;
typedef __attribute__((ext_vector_type(8)))  __bf16 v8bf;
typedef __attribute__((ext_vector_type(8)))  float  v8f;

#define B_  32
#define T_  512
#define I_  128
#define H_  128
#define N_  16
#define HT  2048   // H_*N_

// ============================================================================
// Phase 1: pre = x @ W_ih^T + b_ih + b_hh   -> written into d_out (B*T, HT)
// 128x128 output tile per block, K in chunks of 32, bf16 WMMA / fp32 acc.
// 256 threads = 8 waves; wave w owns rows [16w, 16w+16), all 128 cols.
// ============================================================================
#define LDA 56   // LDS row stride (bf16 elems): 112B = 7*16B aligned chunks
__launch_bounds__(256)
__global__ void rnn_pre_gemm(const float* __restrict__ x,
                             const float* __restrict__ Wih,
                             const float* __restrict__ bih,
                             const float* __restrict__ bhh,
                             float* __restrict__ out)
{
    __shared__ __bf16 Al[128 * LDA];
    __shared__ __bf16 Bl[128 * LDA];

    const int tid  = threadIdx.x;
    const int wave = tid >> 5;
    const int lane = tid & 31;
    const int hi   = lane >> 4;
    const int l16  = lane & 15;

    const int mt = blockIdx.x >> 4;    // 0..127  (row tile over B*T)
    const int nt = blockIdx.x & 15;    // 0..15   (col tile over HT)
    const int m_base = mt * 128;
    const int n_base = nt * 128;

    v8f acc[8] = {};

    const int lrow  = tid >> 1;          // 0..127
    const int lcol0 = (tid & 1) * 16;    // 0 or 16

    for (int kt = 0; kt < 4; ++kt) {
        const int kbase = kt * 32;
        // Stage A (x) and B (W_ih, already K-contiguous) tiles as bf16
        {
            const float* ap = x   + (size_t)(m_base + lrow) * I_ + kbase + lcol0;
            const float* bp = Wih + (size_t)(n_base + lrow) * I_ + kbase + lcol0;
            #pragma unroll
            for (int j = 0; j < 16; j += 4) {
                float4 av = *reinterpret_cast<const float4*>(ap + j);
                float4 bv = *reinterpret_cast<const float4*>(bp + j);
                __bf16* ad = &Al[lrow * LDA + lcol0 + j];
                __bf16* bd = &Bl[lrow * LDA + lcol0 + j];
                ad[0]=(__bf16)av.x; ad[1]=(__bf16)av.y; ad[2]=(__bf16)av.z; ad[3]=(__bf16)av.w;
                bd[0]=(__bf16)bv.x; bd[1]=(__bf16)bv.y; bd[2]=(__bf16)bv.z; bd[3]=(__bf16)bv.w;
            }
        }
        __syncthreads();

        // A fragment (16x32): lane m=l16; lo lanes K=0..15, hi lanes K=16..31
        union { v16bf v; v8bf h[2]; } af;
        const __bf16* arow = &Al[(wave * 16 + l16) * LDA];
        af.h[0] = *reinterpret_cast<const v8bf*>(arow + hi * 16);
        af.h[1] = *reinterpret_cast<const v8bf*>(arow + hi * 16 + 8);

        #pragma unroll
        for (int nb8 = 0; nb8 < 8; ++nb8) {
            // B fragment (32x16): lane n=l16; lo lanes K=0..15, hi lanes K=16..31
            const __bf16* brow = &Bl[(nb8 * 16 + l16) * LDA + hi * 16];
            v16bf bfv = *reinterpret_cast<const v16bf*>(brow);
            acc[nb8] = __builtin_amdgcn_wmma_f32_16x16x32_bf16(
                false, af.v, false, bfv, (short)0, acc[nb8], false, false);
        }
        __syncthreads();
    }

    // Bias + store.  C/D layout: VGPR r -> row (r + 8*hi), col l16.
    #pragma unroll
    for (int nb8 = 0; nb8 < 8; ++nb8) {
        const int col = n_base + nb8 * 16 + l16;
        const float bias = bih[col] + bhh[col];
        #pragma unroll
        for (int r = 0; r < 8; ++r) {
            const int m = m_base + wave * 16 + hi * 8 + r;
            out[(size_t)m * HT + col] = acc[nb8][r] + bias;
        }
    }
}

// ============================================================================
// Phase 2: serial recurrence, in place over d_out.
// One WG per (rnn block, batch-half): 32 WGs, no inter-WG sync needed.
// W block cached once in LDS (bf16); h double-buffered in LDS (bf16);
// weight (B) WMMA fragments are time-invariant -> preloaded in registers.
// pre[t+1] is prefetched into LDS with GLOBAL_LOAD_ASYNC_TO_LDS_B128 so the
// HBM latency overlaps the WMMA chain of step t (serial loop is latency-bound).
// ============================================================================
#define LDW 144  // Wb row stride (bf16): 288B, 16B-aligned chunks
#define LDH 136  // h row stride  (bf16): 272B, 16B-aligned chunks
__launch_bounds__(256)
__global__ void rnn_recurrence(const float* __restrict__ Whh,
                               float* __restrict__ out)
{
    __shared__ __bf16 Wb[128 * LDW];
    __shared__ __bf16 hb[2][16 * LDH];
    __shared__ float  preLds[2][16 * 128];   // double-buffered pre tile (8KB each)

    const int tid  = threadIdx.x;
    const int wave = tid >> 5;       // n-tile 0..7 (16 output cols each)
    const int lane = tid & 31;
    const int hi   = lane >> 4;
    const int l16  = lane & 15;

    const int nb     = blockIdx.x >> 1;        // rnn block 0..15
    const int b_base = (blockIdx.x & 1) * 16;  // batch half

    // --- async prefetch plumbing: each thread owns 8 consecutive pre columns
    const int pf_bl = tid >> 4;            // local batch row 0..15
    const int pf_c  = (tid & 15) * 8;      // column start 0..120
    const size_t pf_base =
        ((size_t)(b_base + pf_bl) * T_) * HT + (size_t)nb * 128 + pf_c;

    // Cache diagonal W block as bf16: Wb[g][k] = Whh[nb*128+g][nb*128+k]
    {
        const int g  = tid >> 1;
        const int c0 = (tid & 1) * 64;
        const float* wp = Whh + (size_t)(nb * 128 + g) * HT + nb * 128 + c0;
        #pragma unroll
        for (int j = 0; j < 64; j += 4) {
            float4 wv = *reinterpret_cast<const float4*>(wp + j);
            __bf16* wd = &Wb[g * LDW + c0 + j];
            wd[0]=(__bf16)wv.x; wd[1]=(__bf16)wv.y; wd[2]=(__bf16)wv.z; wd[3]=(__bf16)wv.w;
        }
    }
    // h0 = 0
    for (int i = tid; i < 2 * 16 * LDH; i += 256)
        (&hb[0][0])[i] = (__bf16)0.0f;

    // Prime the pipeline: async-load pre[t=0] tile into preLds[0].
    {
        unsigned lds0 = (unsigned)(uintptr_t)&preLds[0][pf_bl * 128 + pf_c];
        unsigned long long g0 =
            (unsigned long long)(uintptr_t)(out + pf_base);
        asm volatile("global_load_async_to_lds_b128 %0, %1, off"
                     :: "v"(lds0), "v"(g0) : "memory");
        asm volatile("global_load_async_to_lds_b128 %0, %1, off offset:16"
                     :: "v"(lds0), "v"(g0) : "memory");
        asm volatile("s_wait_asynccnt 0x0" ::: "memory");
    }
    __syncthreads();

    // Preload time-invariant B fragments: out[g] = sum_k h[k]*W[g][k]
    v16bf bf[4];
    #pragma unroll
    for (int kt = 0; kt < 4; ++kt)
        bf[kt] = *reinterpret_cast<const v16bf*>(
            &Wb[(wave * 16 + l16) * LDW + kt * 32 + hi * 16]);

    const int gcol = wave * 16 + l16;   // hidden index within block
    const int col  = nb * 128 + gcol;   // column in HT

    for (int t = 0; t < T_; ++t) {
        const int cur = t & 1, nxt = cur ^ 1;

        // Kick off async prefetch of pre[t+1] -> preLds[nxt] (overlaps WMMA)
        if (t + 1 < T_) {
            unsigned ldsn =
                (unsigned)(uintptr_t)&preLds[nxt][pf_bl * 128 + pf_c];
            unsigned long long gn = (unsigned long long)(uintptr_t)
                (out + pf_base + (size_t)(t + 1) * HT);
            asm volatile("global_load_async_to_lds_b128 %0, %1, off"
                         :: "v"(ldsn), "v"(gn) : "memory");
            asm volatile("global_load_async_to_lds_b128 %0, %1, off offset:16"
                         :: "v"(ldsn), "v"(gn) : "memory");
        }

        // h[t-1] @ Wblock^T : 4 chained bf16 WMMAs (K = 128)
        v8f acc = {};
        #pragma unroll
        for (int kt = 0; kt < 4; ++kt) {
            union { v16bf v; v8bf h[2]; } af;
            const __bf16* arow = &hb[cur][l16 * LDH + kt * 32];
            af.h[0] = *reinterpret_cast<const v8bf*>(arow + hi * 16);
            af.h[1] = *reinterpret_cast<const v8bf*>(arow + hi * 16 + 8);
            acc = __builtin_amdgcn_wmma_f32_16x16x32_bf16(
                false, af.v, false, bf[kt], (short)0, acc, false, false);
        }

        // relu(pre + acc): pre comes from the LDS tile prefetched last step
        #pragma unroll
        for (int r = 0; r < 8; ++r) {
            const int bl = hi * 8 + r;             // local batch 0..15
            const int b  = b_base + bl;
            const size_t idx = ((size_t)b * T_ + t) * HT + col;
            float v = acc[r] + preLds[cur][bl * 128 + gcol];
            v = v > 0.0f ? v : 0.0f;
            out[idx] = v;                          // hs_all[b][t][nb][g]
            hb[nxt][bl * LDH + gcol] = (__bf16)v;
            if (t == T_ - 1)
                out[(size_t)B_ * T_ * HT + (size_t)b * HT + col] = v;  // hidden
        }

        // Publish: wait for this step's async prefetch, then barrier so both
        // the new h tile and the prefetched pre tile are visible to all waves.
        asm volatile("s_wait_asynccnt 0x0" ::: "memory");
        __syncthreads();
    }
}

extern "C" void kernel_launch(void* const* d_in, const int* in_sizes, int n_in,
                              void* d_out, int out_size, void* d_ws, size_t ws_size,
                              hipStream_t stream)
{
    const float* x   = (const float*)d_in[0];
    const float* Wih = (const float*)d_in[1];
    const float* Whh = (const float*)d_in[2];
    const float* bih = (const float*)d_in[3];
    const float* bhh = (const float*)d_in[4];
    float* out = (float*)d_out;

    // pre -> d_out, then serial recurrence in place (+ final hidden copy)
    rnn_pre_gemm<<<dim3(128 * 16), dim3(256), 0, stream>>>(x, Wih, bih, bhh, out);
    rnn_recurrence<<<dim3(32), dim3(256), 0, stream>>>(Whh, out);
}
